// LBLHighwayBiLmV2_10419590660606
// MI455X (gfx1250) — compile-verified, matrix operand order checked by reading, not run on hardware
//
#include <hip/hip_runtime.h>

typedef __bf16 bf16_t;
typedef __attribute__((ext_vector_type(16))) __bf16 v16bf;
typedef __attribute__((ext_vector_type(8)))  float  v8f;
typedef __attribute__((ext_vector_type(4)))  float  f32x4;
typedef __attribute__((ext_vector_type(2)))  unsigned int u32x2;

#define LSEQ  2048
#define BATCH 8
#define HDIM  512
#define WID   16
#define NTAP  17
#define MTILE 64          // M rows per block (2 wave-groups x 2 sub-tiles of 16)

// Scheduling fence: ALU + VMEM may cross (B loads keep pipelining), but
// WMMA and DS reads may NOT -> A fragments stay next to their consumers
// instead of being bulk-hoisted and spilled to scratch.
#if defined(__has_builtin)
#if __has_builtin(__builtin_amdgcn_sched_barrier)
#define SCHED_FENCE() __builtin_amdgcn_sched_barrier(0x77)
#endif
#endif
#ifndef SCHED_FENCE
#define SCHED_FENCE()
#endif

// ---------------- f32 -> bf16 weight conversion ----------------
__global__ void cvt_bf16_kernel(const float* __restrict__ src,
                                bf16_t* __restrict__ dst, int n) {
  int i = blockIdx.x * blockDim.x + threadIdx.x;
  if (i < n) dst[i] = (bf16_t)src[i];
}

// ---------------- per-column softmax tap table [L][17] ----------------
// forward: y[t] = sum_k tab[t][k] * x[t-16+k],  valid k >= 16-t
// backward: y[t] = sum_k tab[t][k] * x[t+k],    valid k <= L-1-t
__global__ void wtab_kernel(const float* __restrict__ w,
                            float* __restrict__ tab, int forward) {
  int t = blockIdx.x * blockDim.x + threadIdx.x;
  if (t >= LSEQ) return;
  float e[NTAP];
  float Z = 0.0f;
  #pragma unroll
  for (int k = 0; k < NTAP; k++) e[k] = expf(w[k]);
  #pragma unroll
  for (int k = 0; k < NTAP; k++) {
    bool valid = forward ? (k >= WID - t) : (k <= (LSEQ - 1 - t));
    if (valid) Z += e[k];
  }
  float inv = 1.0f / Z;
  #pragma unroll
  for (int k = 0; k < NTAP; k++) {
    bool valid = forward ? (k >= WID - t) : (k <= (LSEQ - 1 - t));
    tab[t * NTAP + k] = valid ? e[k] * inv : 0.0f;
  }
}

// ---------------- banded mixing (17-tap conv along L) ----------------
__global__ void band_kernel(const float* __restrict__ X, int xStride,
                            const float* __restrict__ tab,
                            float* __restrict__ Y, int off) {
  const int d = threadIdx.x;            // 0..511
  const int t = blockIdx.x;             // 0..L-1
  const int b = blockIdx.y;             // 0..B-1
  const float* xb = X + (size_t)b * LSEQ * xStride;
  float acc = 0.0f;
  #pragma unroll
  for (int k = 0; k < NTAP; k++) {
    int s = t + off + k;
    float wv = tab[t * NTAP + k];
    if (s >= 0 && s < LSEQ)
      acc = fmaf(wv, xb[(size_t)s * xStride + d], acc);
  }
  Y[((size_t)b * LSEQ + t) * HDIM + d] = acc;
}

// ---------------- highway step: WMMA GEMM + fused epilogue ----------------
// X:   [M][512] f32 current state (row-major)
// W:   [1024][512] bf16 (torch Linear layout: out_features x in_features)
// bias:[1024] f32;   cache: optional residual, row stride cacheStride
// Out: row stride outStride (may alias X when outStride==512; safe because
//      each wave owns a disjoint (row,col) set and reads x_old before writing)
__global__ __launch_bounds__(512) void highway_kernel(
    const float* X,
    const bf16_t* __restrict__ W,
    const float* __restrict__ bias,
    const float* __restrict__ cache, int cacheStride,
    float* Out, int outStride) {
  __shared__ bf16_t abf[MTILE][520];   // bf16 A panel (padded rows)

  const int tid   = threadIdx.x;
  const int mbase = blockIdx.x * MTILE;

  // Stage 64x512 rows as bf16: 8192 float4 reads, 16 per thread.
  #pragma unroll 4
  for (int i = 0; i < 16; i++) {
    int idx4 = tid + 512 * i;          // 0..8191
    int row  = idx4 >> 7;              // 128 float4 per row
    int c4   = idx4 & 127;
    f32x4 v = *reinterpret_cast<const f32x4*>(X + (size_t)(mbase + row) * HDIM + c4 * 4);
    union { bf16_t h[4]; u32x2 u; } p;
    p.h[0] = (bf16_t)v[0];
    p.h[1] = (bf16_t)v[1];
    p.h[2] = (bf16_t)v[2];
    p.h[3] = (bf16_t)v[3];
    *reinterpret_cast<u32x2*>(&abf[row][c4 * 4]) = p.u;   // 8-byte LDS store
  }
  __syncthreads();

  const int wave = tid >> 5;     // 0..15
  const int w8   = wave & 7;     // column partition (same as 8-wave scheme)
  const int wgrp = wave >> 3;    // row half: 0 -> rows 0..31, 1 -> rows 32..63
  const int lane = tid & 31;
  const int g    = lane >> 4;    // lane-group for 16-bit K layout
  const int l16  = lane & 15;    // A: M-row ; B/C: N-column
  const int rbase = wgrp * 32;   // this wave-group's row base within the tile

  union FragU { v16bf v; f32x4 f[2]; };

  for (int nt = 0; nt < 4; nt++) {
    const int nbase = (w8 + nt * 8) * 16;     // [0,512)
    const int col   = nbase + l16;            // output feature (nonlinear half)

    v8f accN[2], accG[2];                     // 2 M-subtiles x {nonlinear,gate}
    const float bN = bias[col];
    const float bG = bias[HDIM + col];
    #pragma unroll
    for (int mt = 0; mt < 2; mt++)
      #pragma unroll
      for (int r = 0; r < 8; r++) { accN[mt][r] = bN; accG[mt][r] = bG; }

    const bf16_t* WrowN = W + (size_t)col * HDIM;           // nonlinear row
    const bf16_t* WrowG = W + (size_t)(HDIM + col) * HDIM;  // gate row

    // Fully unrolled K loop: SSA-chained accumulators, branch-free body.
    #pragma unroll
    for (int kb = 0; kb < 16; kb++) {        // K = 512 in 16 steps of 32
      SCHED_FENCE();                          // keep DS loads in their iteration
      const int ko = kb * 32 + 8 * g;        // lane-group K base
      FragU bn, bg;
      bn.f[0] = *reinterpret_cast<const f32x4*>(WrowN + ko);
      bn.f[1] = *reinterpret_cast<const f32x4*>(WrowN + ko + 16);
      bg.f[0] = *reinterpret_cast<const f32x4*>(WrowG + ko);
      bg.f[1] = *reinterpret_cast<const f32x4*>(WrowG + ko + 16);
      if (kb < 15) {                          // compile-time under full unroll
        __builtin_prefetch(WrowN + ko + 32, 0, 1);
        __builtin_prefetch(WrowG + ko + 32, 0, 1);
      }
      // Reuse B fragments across 2 M-subtiles (low register pressure).
      #pragma unroll
      for (int mt = 0; mt < 2; mt++) {
        FragU a;
        const bf16_t* Arow = &abf[rbase + mt * 16 + l16][0];
        a.f[0] = *reinterpret_cast<const f32x4*>(Arow + ko);
        a.f[1] = *reinterpret_cast<const f32x4*>(Arow + ko + 16);
        accN[mt] = __builtin_amdgcn_wmma_f32_16x16x32_bf16(
            false, a.v, false, bn.v, (short)0, accN[mt], false, false);
        accG[mt] = __builtin_amdgcn_wmma_f32_16x16x32_bf16(
            false, a.v, false, bg.v, (short)0, accG[mt], false, false);
      }
    }
    SCHED_FENCE();

    // Fused highway epilogue: x' = g*x + (1-g)*relu(nl) [+ residual]
    #pragma unroll
    for (int mt = 0; mt < 2; mt++) {
      #pragma unroll
      for (int r = 0; r < 8; r++) {
        const int row = rbase + mt * 16 + r + 8 * g; // C/D: M = r + 8*(lane/16)
        const float nl = accN[mt][r];
        const float gt = accG[mt][r];
        const float gv = 1.0f / (1.0f + expf(-gt));
        const size_t Mrow = (size_t)(mbase + row);
        const float xo = X[Mrow * HDIM + col];
        float xn = gv * xo + (1.0f - gv) * fmaxf(nl, 0.0f);
        if (cache) xn += cache[Mrow * (size_t)cacheStride + col];
        Out[Mrow * (size_t)outStride + col] = xn;
      }
    }
  }
}

extern "C" void kernel_launch(void* const* d_in, const int* in_sizes, int n_in,
                              void* d_out, int out_size, void* d_ws, size_t ws_size,
                              hipStream_t stream) {
  (void)in_sizes; (void)n_in; (void)out_size; (void)ws_size;
  const float* inputs   = (const float*)d_in[0];
  // d_in[1] = masks (all ones; unused by the reference math)
  const float* f_scores = (const float*)d_in[2];
  const float* b_scores = (const float*)d_in[3];
  const float* fW = (const float*)d_in[4];
  const float* fb = (const float*)d_in[5];
  const float* bW = (const float*)d_in[6];
  const float* bb = (const float*)d_in[7];
  float* out = (float*)d_out;

  char* ws = (char*)d_ws;
  const size_t nWelem = 2ull * 2 * 1024 * 512;   // layers*highway*2H*H per direction
  bf16_t* WfB = (bf16_t*)ws;                     // 4 MB
  bf16_t* WbB = WfB + nWelem;                    // 4 MB
  float* wtabF = (float*)(ws + 16ull * 1024 * 1024);
  float* wtabB = wtabF + LSEQ * NTAP;
  float* tmp   = (float*)(ws + 24ull * 1024 * 1024);  // 32 MB state temp

  cvt_bf16_kernel<<<(int)((nWelem + 255) / 256), 256, 0, stream>>>(fW, WfB, (int)nWelem);
  cvt_bf16_kernel<<<(int)((nWelem + 255) / 256), 256, 0, stream>>>(bW, WbB, (int)nWelem);

  const size_t M = (size_t)BATCH * LSEQ;   // 16384 rows
  for (int layer = 0; layer < 2; layer++) {
    for (int dir = 0; dir < 2; dir++) {    // 0 = forward, 1 = backward
      const float* sc = (dir ? b_scores : f_scores) + layer * NTAP;
      float* wtab = dir ? wtabB : wtabF;
      wtab_kernel<<<LSEQ / 128, 128, 0, stream>>>(sc, wtab, dir == 0 ? 1 : 0);

      // Band-mix input: raw inputs for layer 0, previous layer's d_out slice after.
      const float* bx; int bxs;
      if (layer == 0) { bx = inputs; bxs = HDIM; }
      else { bx = out + (size_t)(layer - 1) * M * 1024 + dir * HDIM; bxs = 1024; }
      band_kernel<<<dim3(LSEQ, BATCH), HDIM, 0, stream>>>(bx, bxs, wtab, tmp,
                                                          dir ? 0 : -WID);

      const bf16_t* Wset = dir ? WbB : WfB;
      const float*  bset = dir ? bb  : fb;
      const bf16_t* W0 = Wset + ((size_t)layer * 2 + 0) * 1024 * 512;
      const bf16_t* W1 = Wset + ((size_t)layer * 2 + 1) * 1024 * 512;
      const float*  b0 = bset + (layer * 2 + 0) * 1024;
      const float*  b1 = bset + (layer * 2 + 1) * 1024;

      // Highway step 0: in-place on tmp (element-disjoint waves, read-before-write).
      highway_kernel<<<(int)(M / MTILE), 512, 0, stream>>>(tmp, W0, b0,
                                                           nullptr, 0, tmp, HDIM);
      // Highway step 1: fused residual (+cache for layer>0) and concat/stack
      // write straight into d_out[layer, :, :, dir*H : dir*H+H].
      float* o = out + (size_t)layer * M * 1024 + dir * HDIM;
      const float* cch = layer ? (out + (size_t)(layer - 1) * M * 1024 + dir * HDIM)
                               : nullptr;
      highway_kernel<<<(int)(M / MTILE), 512, 0, stream>>>(tmp, W1, b1,
                                                           cch, 1024, o, 1024);
    }
  }
}
